// Attention_Encoder_61735859913075
// MI455X (gfx1250) — matrix operand order repaired and checked by reading, not compiled, not run on hardware
//
#include <hip/hip_runtime.h>

typedef __attribute__((ext_vector_type(16))) _Float16 v16h;
typedef __attribute__((ext_vector_type(8)))  _Float16 v8h;
typedef __attribute__((ext_vector_type(8)))  float    v8f;
typedef __attribute__((ext_vector_type(4)))  unsigned int v4u;
typedef __attribute__((ext_vector_type(8)))  int      v8i;
typedef __attribute__((ext_vector_type(4)))  int      v4i;

constexpr int Bc = 64, Wc = 32, Tc = 256, Sc = 128, Dc = 64;
constexpr float NEG_INF = -100000.0f;

__global__ __launch_bounds__(256) void attn_enc_kernel(
    const float* __restrict__ q,
    const float* __restrict__ keys,
    const float* __restrict__ vals,
    float* __restrict__ out)
{
    // LDS: 64K stage + 32K Kf16 + 32K Vt(f16, transposed) + 64K probs = 192 KB
    __shared__ __align__(128) float    kStage[Tc * Dc];
    __shared__ __align__(128) _Float16 kH[Tc * Dc];        // [t][d]
    __shared__ __align__(128) _Float16 vT[Dc * Tc];        // [dv][t]
    __shared__ __align__(128) _Float16 probs[8 * 16 * Tc]; // per-wave 16xT

    const int bw   = blockIdx.x;
    const int tid  = threadIdx.x;
    const int lane = tid & 31;
    const int wave = tid >> 5;
    const int m    = lane & 15;   // N / M-within-16 index
    const int hig  = lane >> 4;   // half-wave group

    const float* Kg = keys + (size_t)bw * Tc * Dc;
    const float* Vg = vals + (size_t)bw * Tc * Dc;
    float*       Og = out  + (size_t)bw * Sc * Dc;

    // ---------------- TDM: DMA fp32 K tile (256x64) into LDS ----------------
    if (tid == 0) {
        unsigned long long ga = (unsigned long long)(uintptr_t)Kg;
        unsigned int ldsa = (unsigned int)(uintptr_t)(&kStage[0]);
        v4u g0;
        g0[0] = 1u;                                              // count=1, user desc
        g0[1] = ldsa;                                            // lds_addr
        g0[2] = (unsigned int)(ga & 0xFFFFFFFFu);                // global_addr lo
        g0[3] = (unsigned int)((ga >> 32) & 0x01FFFFFFu)         // global_addr hi
              | (2u << 30);                                      // type=2 (image)
        v8i g1;
        g1[0] = (int)(2u << 16);     // data_size=4B; no multicast/pad/iterate
        g1[1] = (int)(64u << 16);    // tensor_dim0 = 64      (bits [63:48])
        g1[2] = (int)(256u << 16);   // tensor_dim1 = 256     (bits [95:80])
        g1[3] = (int)(64u << 16);    // tile_dim0   = 64      (bits [127:112])
        g1[4] = (int)256;            // tile_dim1   = 256     (bits [143:128])
        g1[5] = 64;                  // tensor_dim0_stride = 64
        g1[6] = 0;
        g1[7] = 0;
        v4i gz4 = {0, 0, 0, 0};
        v8i gz8 = {0, 0, 0, 0, 0, 0, 0, 0};
        __builtin_amdgcn_tensor_load_to_lds(g0, g1, gz4, gz4, gz8, 0);
    }

    // ------------- V: global fp32 -> f16 transposed in LDS (overlaps DMA) ---
#pragma unroll
    for (int i = 0; i < (Tc * Dc) / 256; ++i) {
        int idx = i * 256 + tid;
        float v = Vg[idx];
        int t = idx >> 6, dv = idx & 63;
        vT[dv * Tc + t] = (_Float16)v;
    }

    // ------------- Q fragments (A layout: lane m, k = 16*hig + h) ---------
    const int s0 = wave * 16;
    v16h qa[2];
#pragma unroll
    for (int c = 0; c < 2; ++c) {
        const float* qr = q + (s0 + m) * Dc + c * 32 + hig * 16;
#pragma unroll
        for (int h = 0; h < 8; ++h) qa[c][h]     = (_Float16)qr[h];
#pragma unroll
        for (int h = 0; h < 8; ++h) qa[c][h + 8] = (_Float16)qr[8 + h];
    }

    if (tid == 0) __builtin_amdgcn_s_wait_tensorcnt(0);
    __syncthreads();

    // ------------- K: LDS fp32 -> f16 row-major -----------------------------
#pragma unroll
    for (int i = 0; i < (Tc * Dc) / 256; ++i) {
        int idx = i * 256 + tid;
        kH[idx] = (_Float16)kStage[idx];
    }
    __syncthreads();

    // ------------- Scores: Q[16x64] @ K^T -> 16 tiles of 16x16 (f32 acc) ----
    v8f sc[16];
#pragma unroll
    for (int j = 0; j < 16; ++j) {
        v8f c = {0.f, 0.f, 0.f, 0.f, 0.f, 0.f, 0.f, 0.f};
#pragma unroll
        for (int cc = 0; cc < 2; ++cc) {
            // B[k][n]: lane n = t-in-tile, halves h -> d = cc*32 + h + 16*hig
            const v16h bf = *(const v16h*)(kH + ((j * 16 + m) * Dc + cc * 32 + hig * 16));
            c = __builtin_amdgcn_wmma_f32_16x16x32_f16(false, qa[cc], false, bf,
                                                       (short)0, c, false, false);
        }
        sc[j] = c;
    }

    // ------------- Mask (p==0 -> -INF) + row max ----------------------------
    v8f mx;
#pragma unroll
    for (int r = 0; r < 8; ++r) mx[r] = NEG_INF;
#pragma unroll
    for (int j = 0; j < 16; ++j)
#pragma unroll
        for (int r = 0; r < 8; ++r) {
            float s = sc[j][r];
            s = (s == 0.0f) ? NEG_INF : s;
            sc[j][r] = s;
            mx[r] = fmaxf(mx[r], s);
        }
#pragma unroll
    for (int off = 1; off < 16; off <<= 1)
#pragma unroll
        for (int r = 0; r < 8; ++r)
            mx[r] = fmaxf(mx[r], __shfl_xor(mx[r], off, 16));

    // ------------- exp + row sum --------------------------------------------
    v8f sum = {0.f, 0.f, 0.f, 0.f, 0.f, 0.f, 0.f, 0.f};
#pragma unroll
    for (int j = 0; j < 16; ++j)
#pragma unroll
        for (int r = 0; r < 8; ++r) {
            float e = exp2f((sc[j][r] - mx[r]) * 1.44269504088896f);
            sc[j][r] = e;
            sum[r] += e;
        }
#pragma unroll
    for (int off = 1; off < 16; off <<= 1)
#pragma unroll
        for (int r = 0; r < 8; ++r)
            sum[r] += __shfl_xor(sum[r], off, 16);

    // ------------- C-layout -> A-layout transpose via wave-private LDS ------
    _Float16* pb = probs + wave * 16 * Tc;
#pragma unroll
    for (int j = 0; j < 16; ++j)
#pragma unroll
        for (int r = 0; r < 8; ++r)
            pb[(r + 8 * hig) * Tc + j * 16 + m] = (_Float16)sc[j][r];

    // ------------- attn[16x256] @ V[256x64] ---------------------------------
    v8f acc[4];
#pragma unroll
    for (int jn = 0; jn < 4; ++jn)
#pragma unroll
        for (int r = 0; r < 8; ++r) acc[jn][r] = 0.f;

#pragma unroll
    for (int kb = 0; kb < 8; ++kb) {
        const int t0 = kb * 32;
        // A fragment: lane row m, halves h -> t = t0 + h + 16*hig
        v8h alo = *(const v8h*)(pb + m * Tc + t0 + hig * 16);
        v8h ahi = *(const v8h*)(pb + m * Tc + t0 + 8 + hig * 16);
        v16h af = __builtin_shufflevector(alo, ahi, 0, 1, 2, 3, 4, 5, 6, 7,
                                          8, 9, 10, 11, 12, 13, 14, 15);
#pragma unroll
        for (int jn = 0; jn < 4; ++jn) {
            // B fragment from transposed V: lane n = dv, halves h -> t = t0 + h + 16*hig
            const v16h bf = *(const v16h*)(vT + (jn * 16 + m) * Tc + t0 + hig * 16);
            acc[jn] = __builtin_amdgcn_wmma_f32_16x16x32_f16(false, af, false, bf,
                                                             (short)0, acc[jn], false, false);
        }
    }

    // ------------- normalize + store ----------------------------------------
    v8f inv;
#pragma unroll
    for (int r = 0; r < 8; ++r) inv[r] = 1.0f / sum[r];
#pragma unroll
    for (int jn = 0; jn < 4; ++jn)
#pragma unroll
        for (int r = 0; r < 8; ++r)
            Og[(s0 + r + 8 * hig) * Dc + jn * 16 + m] = acc[jn][r] * inv[r];
}

extern "C" void kernel_launch(void* const* d_in, const int* in_sizes, int n_in,
                              void* d_out, int out_size, void* d_ws, size_t ws_size,
                              hipStream_t stream) {
    const float* q = (const float*)d_in[0];
    const float* k = (const float*)d_in[1];
    const float* v = (const float*)d_in[2];
    float* o = (float*)d_out;
    dim3 grid(Bc * Wc), block(256);
    hipLaunchKernelGGL(attn_enc_kernel, grid, block, 0, stream, q, k, v, o);
}
